// _RPN_FPN_4733053960295
// MI455X (gfx1250) — matrix operand order repaired and checked by reading, not compiled, not run on hardware
//
#include <hip/hip_runtime.h>
#include <hip/hip_bf16.h>
#include <math.h>

typedef unsigned int   u32;
typedef unsigned short u16;
typedef unsigned long long u64;
typedef __bf16 v16bf __attribute__((ext_vector_type(16)));
typedef float  v8f   __attribute__((ext_vector_type(8)));
typedef u32    u32x4 __attribute__((ext_vector_type(4)));
typedef u32    u32x8 __attribute__((ext_vector_type(8)));
typedef float  f32x4 __attribute__((ext_vector_type(4)));

#define NLVL 5
#define NA_TOTAL 261888      // 3 * (256^2+128^2+64^2+32^2+16^2)
#define PRE_NMS 1000
#define POST_NMS 300
#define KCH 72               // K=2304 in chunks of 32
#define NTILE 32             // 512 out channels / 16
#define TILES_TOTAL 2728     // sum(S/32)
#define BATCH 2
#define APIX 2312            // per-pixel A stride in u16 (2304 + pad)
#define H1STRIDE 516         // per-pixel h1 stride in f32 (512 + pad)
#define CONV_LDS 147968      // 32 * APIX * 2 bytes
#define CONV_THREADS 512     // 16 waves: each owns 32 output channels

struct LvlTab {
  int S[NLVL]; int W[NLVL]; int H[NLVL];
  int tileBase[NLVL]; int stride[NLVL]; int anchorBase[NLVL];
  long long featOff[NLVL];
};

__device__ __forceinline__ u16 f2bf(float f) {
  u32 u = __float_as_uint(f);
  u32 r = u + 0x7FFFu + ((u >> 16) & 1u);
  return (u16)(r >> 16);
}
__device__ __forceinline__ u32 skey(float f) {   // monotone float->uint map
  u32 u = __float_as_uint(f);
  return (u & 0x80000000u) ? ~u : (u | 0x80000000u);
}

// ---------- 1. NCHW f32 -> NHWC bf16 (tiled transpose) ----------
__global__ __launch_bounds__(1024, 1)
void k_convert(const float* __restrict__ in, u16* __restrict__ out, int S) {
  __shared__ float tilebuf[32][33];
  int b = blockIdx.z;
  int c0 = blockIdx.y << 5;
  int s0 = blockIdx.x << 5;
  int tx = threadIdx.x, ty = threadIdx.y;
  tilebuf[ty][tx] = in[((long long)b * 256 + c0 + ty) * S + s0 + tx];
  __syncthreads();
  out[((long long)b * S + s0 + ty) * 256 + c0 + tx] = f2bf(tilebuf[tx][ty]);
}

// ---------- 2. pack conv weights into WMMA B-fragment order ----------
__global__ void k_packw(const float* __restrict__ cw, u16* __restrict__ gw) {
  int t = blockIdx.x * blockDim.x + threadIdx.x;
  if (t >= NTILE * KCH * 32 * 16) return;
  int e    = t & 15;
  int lane = (t >> 4) & 31;
  int kc   = (t >> 9) % KCH;
  int nt   = t / (512 * KCH);
  int N    = nt * 16 + (lane & 15);
  int kin  = ((lane >> 4) << 3) + (e & 7) + ((e >> 3) << 4);
  int K    = kc * 32 + kin;
  int tap  = K >> 8;          // K = tap*256 + c
  int c    = K & 255;
  gw[t] = f2bf(cw[(N * 256 + c) * 9 + tap]);
}

// ---------- 3. pack 1x1 head weights / biases ----------
__global__ void k_packpw(const float* __restrict__ cls_w, const float* __restrict__ bbox_w,
                         const float* __restrict__ cls_b, const float* __restrict__ bbox_b,
                         float* __restrict__ pw, float* __restrict__ pb) {
  int t = blockIdx.x * blockDim.x + threadIdx.x;
  if (t < 18 * 512) {
    int o = t >> 9, c = t & 511;
    pw[t] = (o < 6) ? cls_w[o * 512 + c] : bbox_w[(o - 6) * 512 + c];
  } else if (t < 18 * 512 + 18) {
    int o = t - 18 * 512;
    pb[o] = (o < 6) ? cls_b[o] : bbox_b[o - 6];
  }
}

// ---------- 4. fused conv3x3 (WMMA) + heads + decode ----------
// 16 waves; wave wv owns N-tiles (2*wv, 2*wv+1) and both 16-pixel M-subtiles
// => 4 independent WMMA accumulator chains per wave (hides XDL RAW hazards).
__global__ __launch_bounds__(CONV_THREADS, 1)
void k_conv_fused(const u16* __restrict__ feat, const u16* __restrict__ gw,
                  const float* __restrict__ pw, const float* __restrict__ pb,
                  const float* __restrict__ conv_b, const float* __restrict__ im_info,
                  float* __restrict__ scores, float* __restrict__ boxes, LvlTab tab) {
  extern __shared__ char smem[];
  u16*   As = (u16*)smem;              // [32 px][APIX u16] im2col tile
  float* h1 = (float*)smem;            // aliased after WMMA: [32 px][H1STRIDE f32]
  float* lg = (float*)(smem + 32 * H1STRIDE * 4);  // [32 px][18] logits

  int blk = blockIdx.x;
  int b = blk / TILES_TOTAL;
  int t = blk - b * TILES_TOTAL;
  int lvl = 0;
#pragma unroll
  for (int l = 1; l < NLVL; ++l) if (t >= tab.tileBase[l]) lvl = l;
  int tileIn = t - tab.tileBase[lvl];

  int tid = threadIdx.x;
  int lane = tid & 31;
  int wv = tid >> 5;                   // 0..15
  int S = tab.S[lvl], W = tab.W[lvl], H = tab.H[lvl];
  const u16* fb = feat + tab.featOff[lvl] + (long long)b * S * 256;

  // stage im2col A tile: 288 (pixel,tap) jobs over 16 waves
  for (int j = wv; j < 288; j += 16) {
    int p = j & 31;
    int tap = j >> 5;
    int m = tileIn * 32 + p;
    int y = m / W, x = m - y * W;
    int ky = tap / 3, kx = tap - ky * 3;
    int sy = y + ky - 1, sx = x + kx - 1;
    u32x4 v = {0u, 0u, 0u, 0u};
    if (sy >= 0 && sy < H && sx >= 0 && sx < W)
      v = *(const u32x4*)(fb + (long long)(sy * W + sx) * 256 + lane * 8);
    *(u32x4*)(As + p * APIX + tap * 256 + lane * 8) = v;
  }
  __syncthreads();

  // implicit GEMM: wave -> 32 pixels x 32 channels, K = 2304
  v8f acc00 = {0.f, 0.f, 0.f, 0.f, 0.f, 0.f, 0.f, 0.f};
  v8f acc01 = acc00, acc10 = acc00, acc11 = acc00;
  int hsel = lane >> 4;
  int pA = lane & 15;
  const u16* a0p = As + pA * APIX + hsel * 8;
  const u16* a1p = a0p + 16 * APIX;
  const u32x8* gwp0 = ((const u32x8*)gw) + (long long)(wv * 2) * KCH * 32 + lane;
  const u32x8* gwp1 = gwp0 + (long long)KCH * 32;

  for (int kc = 0; kc < KCH; ++kc) {
    u32x8 bw0 = gwp0[kc * 32];
    u32x8 bw1 = gwp1[kc * 32];
    u32x4 lo0 = *(const u32x4*)(a0p + kc * 32);
    u32x4 hi0 = *(const u32x4*)(a0p + kc * 32 + 16);
    u32x4 lo1 = *(const u32x4*)(a1p + kc * 32);
    u32x4 hi1 = *(const u32x4*)(a1p + kc * 32 + 16);
    v16bf amat0 = __builtin_bit_cast(v16bf, __builtin_shufflevector(lo0, hi0, 0,1,2,3,4,5,6,7));
    v16bf amat1 = __builtin_bit_cast(v16bf, __builtin_shufflevector(lo1, hi1, 0,1,2,3,4,5,6,7));
    v16bf bmat0 = __builtin_bit_cast(v16bf, bw0);
    v16bf bmat1 = __builtin_bit_cast(v16bf, bw1);
    // 4 independent chains -> no forced v_nop bubbles between WMMAs
    acc00 = __builtin_amdgcn_wmma_f32_16x16x32_bf16(false, amat0, false, bmat0, (short)0, acc00, false, false);
    acc01 = __builtin_amdgcn_wmma_f32_16x16x32_bf16(false, amat0, false, bmat1, (short)0, acc01, false, false);
    acc10 = __builtin_amdgcn_wmma_f32_16x16x32_bf16(false, amat1, false, bmat0, (short)0, acc10, false, false);
    acc11 = __builtin_amdgcn_wmma_f32_16x16x32_bf16(false, amat1, false, bmat1, (short)0, acc11, false, false);
    if (kc + 1 < KCH) {
      __builtin_prefetch(gwp0 + (kc + 1) * 32, 0, 3);  // locality 3 -> near (WGP) scope
      __builtin_prefetch(gwp1 + (kc + 1) * 32, 0, 3);
    }
  }
  __syncthreads();  // everyone done reading As -> safe to alias h1

  // bias + ReLU -> h1 tile in LDS (never spilled to HBM)
  {
    int n0 = wv * 32 + (lane & 15);
    int n1 = n0 + 16;
    float bias0 = conv_b[n0];
    float bias1 = conv_b[n1];
    int prow = hsel * 8;
#pragma unroll
    for (int v = 0; v < 8; ++v) {
      float z;
      z = acc00[v] + bias0; h1[(v + prow) * H1STRIDE + n0]      = z > 0.f ? z : 0.f;
      z = acc01[v] + bias1; h1[(v + prow) * H1STRIDE + n1]      = z > 0.f ? z : 0.f;
      z = acc10[v] + bias0; h1[(v + prow + 16) * H1STRIDE + n0] = z > 0.f ? z : 0.f;
      z = acc11[v] + bias1; h1[(v + prow + 16) * H1STRIDE + n1] = z > 0.f ? z : 0.f;
    }
  }
  __syncthreads();

  // 1x1 heads: 32 px * 18 outputs, dot over 512
  for (int job = tid; job < 576; job += CONV_THREADS) {
    int px = job / 18, o = job - (job / 18) * 18;
    float s = pb[o];
    const float* hp = h1 + px * H1STRIDE;
    const float* wp = pw + o * 512;
#pragma unroll 4
    for (int c = 0; c < 512; c += 4) {
      f32x4 hv = *(const f32x4*)(hp + c);
      f32x4 wv2 = *(const f32x4*)(wp + c);
      s += hv.x * wv2.x + hv.y * wv2.y + hv.z * wv2.z + hv.w * wv2.w;
    }
    lg[px * 18 + o] = s;
  }
  __syncthreads();

  // softmax + anchor decode + clip, write scores/boxes
  if (tid < 96) {
    int px = tid / 3, a = tid - (tid / 3) * 3;
    int m = tileIn * 32 + px;
    int y = m / W, x = m - y * W;
    const float* l = lg + px * 18;
    float scv = 1.f / (1.f + __expf(l[a] - l[3 + a]));   // p(class1)
    float dx = l[6 + 4 * a], dy = l[7 + 4 * a], dwv = l[8 + 4 * a], dhv = l[9 + 4 * a];
    float st = (float)tab.stride[lvl];
    float rr = (a == 0) ? 0.5f : ((a == 1) ? 1.f : 2.f);
    float wsr = roundf(sqrtf(st * st / rr));
    float wsc = wsr * 8.f;
    float hsc = roundf(wsr * rr) * 8.f;
    float ctr = (st - 1.f) * 0.5f + 0.5f;
    float acx = ctr + (float)x * st;
    float acy = ctr + (float)y * st;
    float pcx = dx * wsc + acx, pcy = dy * hsc + acy;
    float pwd = __expf(dwv) * wsc, phd = __expf(dhv) * hsc;
    float imh = im_info[b * 3 + 0], imw = im_info[b * 3 + 1];
    float x1 = fminf(fmaxf(pcx - 0.5f * pwd, 0.f), imw - 1.f);
    float y1 = fminf(fmaxf(pcy - 0.5f * phd, 0.f), imh - 1.f);
    float x2 = fminf(fmaxf(pcx + 0.5f * pwd, 0.f), imw - 1.f);
    float y2 = fminf(fmaxf(pcy + 0.5f * phd, 0.f), imh - 1.f);
    long long gi = (long long)b * NA_TOTAL + tab.anchorBase[lvl] + m * 3 + a;
    scores[gi] = scv;
    float* bo = boxes + gi * 4;
    bo[0] = x1; bo[1] = y1; bo[2] = x2; bo[3] = y2;
  }
}

// ---------- 5. top-1000 per batch: radix select + bitonic sort ----------
__global__ __launch_bounds__(1024, 1)
void k_topk(const float* __restrict__ scores, const float* __restrict__ boxes,
            float* __restrict__ topbox) {
  __shared__ u32 hist[256];
  __shared__ u32 shPrefix, shRem, shCnt;
  __shared__ u64 sk[2048];
  int b = blockIdx.x, tid = threadIdx.x;
  const float* sb = scores + (long long)b * NA_TOTAL;
  if (tid == 0) { shPrefix = 0u; shRem = PRE_NMS; shCnt = 0u; }
  for (int t2 = tid; t2 < 2048; t2 += 1024) sk[t2] = 0ull;
  __syncthreads();
  for (int r = 0; r < 4; ++r) {
    int shift = 24 - 8 * r;
    u32 himask = (r == 0) ? 0u : (0xFFFFFFFFu << (shift + 8));
    if (tid < 256) hist[tid] = 0u;
    __syncthreads();
    u32 pfx = shPrefix;
    for (int i = tid; i < NA_TOTAL; i += 1024) {
      u32 k = skey(sb[i]);
      if ((k & himask) == (pfx & himask)) atomicAdd(&hist[(k >> shift) & 255u], 1u);
    }
    __syncthreads();
    if (tid == 0) {
      u32 rem = shRem;
      for (int v2 = 255; v2 >= 0; --v2) {
        u32 c = hist[v2];
        if (c < rem) rem -= c;
        else { shPrefix = pfx | ((u32)v2 << shift); shRem = rem; break; }
      }
    }
    __syncthreads();
  }
  u32 T = shPrefix;
  for (int i = tid; i < NA_TOTAL; i += 1024) {
    u32 k = skey(sb[i]);
    if (k >= T) {
      u32 pos = atomicAdd(&shCnt, 1u);
      if (pos < 2048u) sk[pos] = ((u64)k << 32) | (u32)(~(u32)i);  // key desc, idx asc
    }
  }
  __syncthreads();
  for (u32 kk = 2; kk <= 2048; kk <<= 1) {
    for (u32 j = kk >> 1; j > 0; j >>= 1) {
      __syncthreads();
      for (u32 t2 = tid; t2 < 2048; t2 += 1024) {
        u32 ixj = t2 ^ j;
        if (ixj > t2) {
          u64 aa = sk[t2], bb = sk[ixj];
          bool up = (t2 & kk) == 0;                 // descending blocks
          if (up ? (aa < bb) : (aa > bb)) { sk[t2] = bb; sk[ixj] = aa; }
        }
      }
    }
  }
  __syncthreads();
  if (tid < PRE_NMS) {
    u64 e2 = sk[tid];
    u32 kv = (u32)(e2 >> 32);
    u32 idx = ~(u32)(e2 & 0xFFFFFFFFull);
    float x1 = 0.f, y1 = 0.f, x2 = 0.f, y2 = 0.f;
    if (kv != 0u) {
      const float* bp = boxes + ((long long)b * NA_TOTAL + idx) * 4;
      x1 = bp[0]; y1 = bp[1]; x2 = bp[2]; y2 = bp[3];
    }
    float* op = topbox + ((long long)b * PRE_NMS + tid) * 4;
    op[0] = x1; op[1] = y1; op[2] = x2; op[3] = y2;
  }
}

// ---------- 6. NMS + roi scatter ----------
__global__ __launch_bounds__(1024, 1)
void k_nms(const float* __restrict__ topbox, float* __restrict__ out) {
  __shared__ float bx[PRE_NMS * 4];
  __shared__ float ar[PRE_NMS];
  __shared__ u32 kp[PRE_NMS];
  __shared__ u32 sc[1024];
  int b = blockIdx.x, tid = threadIdx.x;
  if (tid < PRE_NMS) {
    const float* bp = topbox + ((long long)b * PRE_NMS + tid) * 4;
    float x1 = bp[0], y1 = bp[1], x2 = bp[2], y2 = bp[3];
    bx[tid * 4 + 0] = x1; bx[tid * 4 + 1] = y1; bx[tid * 4 + 2] = x2; bx[tid * 4 + 3] = y2;
    ar[tid] = (x2 - x1 + 1.f) * (y2 - y1 + 1.f);
    kp[tid] = 1u;
  }
  __syncthreads();
  for (int i = 0; i < PRE_NMS; ++i) {
    if (kp[i] && tid > i && tid < PRE_NMS) {
      float iw = fminf(bx[i * 4 + 2], bx[tid * 4 + 2]) - fmaxf(bx[i * 4 + 0], bx[tid * 4 + 0]) + 1.f;
      float ih = fminf(bx[i * 4 + 3], bx[tid * 4 + 3]) - fmaxf(bx[i * 4 + 1], bx[tid * 4 + 1]) + 1.f;
      iw = iw > 0.f ? iw : 0.f; ih = ih > 0.f ? ih : 0.f;
      float inter = iw * ih;
      float iou = inter / (ar[i] + ar[tid] - inter);
      if (iou > 0.7f) kp[tid] = 0u;
    }
    __syncthreads();
  }
  u32 myk = (tid < PRE_NMS) ? kp[tid] : 0u;
  sc[tid] = myk;
  __syncthreads();
  for (int off = 1; off < 1024; off <<= 1) {
    u32 add = (tid >= off) ? sc[tid - off] : 0u;
    __syncthreads();
    sc[tid] += add;
    __syncthreads();
  }
  if (tid < PRE_NMS && myk) {
    u32 rank = sc[tid] - 1u;
    if (rank < POST_NMS) {
      float* op = out + ((long long)b * POST_NMS + rank) * 5;
      op[0] = (float)b;
      op[1] = bx[tid * 4 + 0]; op[2] = bx[tid * 4 + 1];
      op[3] = bx[tid * 4 + 2]; op[4] = bx[tid * 4 + 3];
    }
  }
}

extern "C" void kernel_launch(void* const* d_in, const int* in_sizes, int n_in,
                              void* d_out, int out_size, void* d_ws, size_t ws_size,
                              hipStream_t stream) {
  (void)in_sizes; (void)n_in; (void)ws_size;
  const float* feats_in[NLVL] = {
    (const float*)d_in[0], (const float*)d_in[1], (const float*)d_in[2],
    (const float*)d_in[3], (const float*)d_in[4]};
  const float* im_info = (const float*)d_in[5];
  const float* conv_w  = (const float*)d_in[8];
  const float* conv_b  = (const float*)d_in[9];
  const float* cls_w   = (const float*)d_in[10];
  const float* cls_b   = (const float*)d_in[11];
  const float* bbox_w  = (const float*)d_in[12];
  const float* bbox_b  = (const float*)d_in[13];

  static const int Hs[NLVL] = {256, 128, 64, 32, 16};
  LvlTab tab;
  long long featOffE[NLVL];
  long long accE = 0;
  int tb = 0, ab = 0;
  for (int l = 0; l < NLVL; ++l) {
    int H = Hs[l], W = Hs[l], S = H * W;
    tab.S[l] = S; tab.W[l] = W; tab.H[l] = H;
    tab.stride[l] = 4 << l;
    tab.tileBase[l] = tb;   tb += S / 32;
    tab.anchorBase[l] = ab; ab += S * 3;
    featOffE[l] = accE; tab.featOff[l] = accE;
    accE += (long long)BATCH * S * 256;
  }

  char* wsb = (char*)d_ws;
  size_t cur = 0;
  auto salloc = [&](size_t bytes) { size_t o = cur; cur += (bytes + 255) & ~(size_t)255; return o; };
  u16*   feat   = (u16*)  (wsb + salloc((size_t)accE * 2));
  u16*   gw     = (u16*)  (wsb + salloc((size_t)NTILE * KCH * 32 * 16 * 2));
  float* pw     = (float*)(wsb + salloc(18 * 512 * 4));
  float* pb     = (float*)(wsb + salloc(18 * 4));
  float* scores = (float*)(wsb + salloc((size_t)BATCH * NA_TOTAL * 4));
  float* boxes  = (float*)(wsb + salloc((size_t)BATCH * NA_TOTAL * 16));
  float* topbox = (float*)(wsb + salloc((size_t)BATCH * PRE_NMS * 16));

  for (int l = 0; l < NLVL; ++l) {
    int S = tab.S[l];
    dim3 g(S / 32, 8, BATCH), blk(32, 32, 1);
    k_convert<<<g, blk, 0, stream>>>(feats_in[l], feat + featOffE[l], S);
  }
  {
    int n = NTILE * KCH * 32 * 16;
    k_packw<<<(n + 255) / 256, 256, 0, stream>>>(conv_w, gw);
  }
  k_packpw<<<(18 * 512 + 18 + 255) / 256, 256, 0, stream>>>(cls_w, bbox_w, cls_b, bbox_b, pw, pb);

  hipFuncSetAttribute((const void*)k_conv_fused,
                      hipFuncAttributeMaxDynamicSharedMemorySize, CONV_LDS);
  k_conv_fused<<<BATCH * TILES_TOTAL, CONV_THREADS, CONV_LDS, stream>>>(
      feat, gw, pw, pb, conv_b, im_info, scores, boxes, tab);

  k_topk<<<BATCH, 1024, 0, stream>>>(scores, boxes, topbox);
  hipMemsetAsync(d_out, 0, (size_t)out_size * sizeof(float), stream);
  k_nms<<<BATCH, 1024, 0, stream>>>(topbox, (float*)d_out);
}
